// MambaICL_25855703121983
// MI455X (gfx1250) — compile-verified, matrix operand order checked by reading, not run on hardware
//
#include <hip/hip_runtime.h>
#include <cstdint>

typedef _Float16 v16h __attribute__((ext_vector_type(16)));
typedef float    v8f  __attribute__((ext_vector_type(8)));

#define NL      12
#define INDIM   230
#define KIN     256      // input dim padded to x32
#define DM      256      // d_model
#define DI      512      // d_inner
#define DSTATE  16
#define DTR     16       // dt_rank
#define NX      48       // DTR + 2*DSTATE
#define KDT     32       // dt_rank padded to x32

#define WIT_ELEMS  (DM * KIN)        // 65536   WiT[n][k]
#define INP_ELEMS  (2 * DI * DM)     // 262144  inpT[n<1024][k<256]
#define XPR_ELEMS  (NX * DI)         // 24576   xprT[n<48][k<512]
#define DTT_ELEMS  (DI * KDT)        // 16384   dtT [n<512][k<32]
#define OUTP_ELEMS (DM * DI)         // 131072  outT[n<256][k<512]
#define LAYER_ELEMS (INP_ELEMS + XPR_ELEMS + DTT_ELEMS + OUTP_ELEMS)

// ---------------------------------------------------------------------------
// Tile loader: WMMA A (16x32 rows) and B (32x16 cols) share one per-lane
// layout when the source is [row-or-col][K] with K contiguous (CDNA5 ISA
// 5.7.12.2, 16-bit A-matrix table):
//   lane L: row/col = base + (L & 15)
//   lo lanes: K = {0..7, 16..23};  hi lanes: K = {8..15, 24..31}
// ---------------------------------------------------------------------------
__device__ __forceinline__ v16h load_tile(const _Float16* base, int ldk,
                                          int rcbase, int k0, int lane) {
  const int r  = rcbase + (lane & 15);
  const int hi = (lane >> 4) & 1;
  const _Float16* p = base + (size_t)r * ldk + k0 + hi * 8;
  union { v16h v; uint32_t u[8]; } t;
#pragma unroll
  for (int i = 0; i < 8; ++i) {
    const int kd = ((i & 3) << 1) + ((i >> 2) << 4);
    t.u[i] = *reinterpret_cast<const uint32_t*>(p + kd);
  }
  return t.v;
}

__device__ __forceinline__ v8f wmma16(v16h a, v16h b, v8f c) {
  return __builtin_amdgcn_wmma_f32_16x16x32_f16(false, a, false, b,
                                                (short)0, c, false, false);
}

__device__ __forceinline__ float silu_f(float x) { return x / (1.f + __expf(-x)); }
__device__ __forceinline__ float softplus_f(float x) {
  return (x > 20.f) ? x : __logf(1.f + __expf(x));
}

// Async global->LDS copy (16B), tracked by ASYNCcnt.  Generic pointers to
// __shared__ carry the LDS byte offset in their low 32 bits (ISA 10.2).
__device__ __forceinline__ void async_b128(void* lds_ptr, const void* gptr) {
  uint32_t lo = (uint32_t)(uintptr_t)lds_ptr;
  unsigned long long ga = (unsigned long long)(uintptr_t)gptr;
  asm volatile("global_load_async_to_lds_b128 %0, %1, off"
               :: "v"(lo), "v"(ga) : "memory");
}
__device__ __forceinline__ void wait_async0() {
  asm volatile("s_wait_asynccnt 0x0" ::: "memory");
}

// ---------------------------------------------------------------------------
// Weight prep: transpose to [N][K] f16, pad K to x32.
// ---------------------------------------------------------------------------
__global__ void prep_wiT(const float* __restrict__ Wi, _Float16* __restrict__ ws) {
  int i = blockIdx.x * 256 + threadIdx.x;        // 65536
  int n = i >> 8, k = i & 255;
  ws[i] = (_Float16)((k < INDIM) ? Wi[(size_t)k * DM + n] : 0.f);
}
__global__ void prep_inpT(const float* __restrict__ W, _Float16* __restrict__ ws) {
  int l = blockIdx.y;
  int i = blockIdx.x * 256 + threadIdx.x;        // 262144 per layer
  int n = i >> 8, k = i & 255;
  ws[WIT_ELEMS + (size_t)l * LAYER_ELEMS + i] =
      (_Float16)W[(size_t)l * DM * 2 * DI + (size_t)k * 2 * DI + n];
}
__global__ void prep_xprT(const float* __restrict__ W, _Float16* __restrict__ ws) {
  int l = blockIdx.y;
  int i = blockIdx.x * 256 + threadIdx.x;        // 24576 per layer
  int n = i >> 9, k = i & 511;
  ws[WIT_ELEMS + (size_t)l * LAYER_ELEMS + INP_ELEMS + i] =
      (_Float16)W[(size_t)l * DI * NX + (size_t)k * NX + n];
}
__global__ void prep_dtT(const float* __restrict__ W, _Float16* __restrict__ ws) {
  int l = blockIdx.y;
  int i = blockIdx.x * 256 + threadIdx.x;        // 16384 per layer
  int n = i >> 5, k = i & 31;
  float v = (k < DTR) ? W[(size_t)l * DTR * DI + (size_t)k * DI + n] : 0.f;
  ws[WIT_ELEMS + (size_t)l * LAYER_ELEMS + INP_ELEMS + XPR_ELEMS + i] = (_Float16)v;
}
__global__ void prep_outT(const float* __restrict__ W, _Float16* __restrict__ ws) {
  int l = blockIdx.y;
  int i = blockIdx.x * 256 + threadIdx.x;        // 131072 per layer
  int n = i >> 9, k = i & 511;
  ws[WIT_ELEMS + (size_t)l * LAYER_ELEMS + INP_ELEMS + XPR_ELEMS + DTT_ELEMS + i] =
      (_Float16)W[(size_t)l * DI * DM + (size_t)k * DM + n];
}

// ---------------------------------------------------------------------------
// Fully fused 12-layer Mamba (L==1: scan collapses to
//   y = xc * (softplus(dt)*dot(B,C) + D) * silu(z); dA/A_log unused).
// One workgroup = 16 batch rows persistent in LDS across all layers.
// x_proj / dt_proj weights for each layer stream into LDS via async-to-LDS
// copies overlapped with GEMM1's WMMAs.
// ---------------------------------------------------------------------------
__global__ __launch_bounds__(256) void mamba_fused(
    const float* __restrict__ x,       const float* __restrict__ bi,
    const float* __restrict__ conv_w,  const float* __restrict__ conv_b,
    const float* __restrict__ dt_b,    const float* __restrict__ Dp,
    const float* __restrict__ Wo,      const float* __restrict__ bo,
    const _Float16* __restrict__ ws,   float* __restrict__ out) {
  __shared__ _Float16 sH[16 * DM];      // running hidden state, f16      8 KB
  __shared__ _Float16 sXC[16 * DI];     // silu(conv(xr)); input staging 16 KB
  __shared__ _Float16 sZ[16 * DI];      // silu(z)                       16 KB
  __shared__ _Float16 sY[16 * DI];      // gated output                  16 KB
  __shared__ _Float16 sXPR[XPR_ELEMS];  // async-staged x_proj weights   48 KB
  __shared__ _Float16 sDTT[DTT_ELEMS];  // async-staged dt_proj weights  32 KB
  __shared__ float    sDBL[16 * NX];    // [dt | B | C] f32               3 KB
  __shared__ _Float16 sDtA[16 * KDT];   // dt rows padded K 16->32        1 KB
  __shared__ float    sBC[16];          // per-row dot(B, C)

  const int tid  = threadIdx.x;
  const int wave = tid >> 5;
  const int lane = tid & 31;
  const int hi   = (lane >> 4) & 1;
  const int row0 = blockIdx.x * 16;

  // stage 16 input rows (pad 230 -> 256) as f16 A-tile
  for (int i = tid; i < 16 * KIN; i += 256) {
    int r = i >> 8, c = i & 255;
    float v = (c < INDIM) ? x[(size_t)(row0 + r) * INDIM + c] : 0.f;
    sXC[i] = (_Float16)v;
  }
  __syncthreads();

  // GEMM0: h = x @ Wi + bi  (16 n-tiles, dual-tile per wave)
  {
    int t0 = wave * 2, t1 = t0 + 1;
    v8f acc0 = {}, acc1 = {};
#pragma unroll
    for (int k = 0; k < KIN / 32; ++k) {
      v16h a  = load_tile(sXC, KIN, 0, k * 32, lane);
      v16h b0 = load_tile(ws, KIN, t0 * 16, k * 32, lane);
      v16h b1 = load_tile(ws, KIN, t1 * 16, k * 32, lane);
      acc0 = wmma16(a, b0, acc0);
      acc1 = wmma16(a, b1, acc1);
    }
    int c0 = t0 * 16 + (lane & 15), c1 = c0 + 16;
    float b0 = bi[c0], b1 = bi[c1];
#pragma unroll
    for (int v = 0; v < 8; ++v) {
      int m = v + hi * 8;
      sH[m * DM + c0] = (_Float16)(acc0[v] + b0);
      sH[m * DM + c1] = (_Float16)(acc1[v] + b1);
    }
  }
  __syncthreads();

  for (int l = 0; l < NL; ++l) {
    const _Float16* inpT = ws + WIT_ELEMS + (size_t)l * LAYER_ELEMS;
    const _Float16* xprT = inpT + INP_ELEMS;
    const _Float16* dtT  = xprT + XPR_ELEMS;
    const _Float16* outT = dtT + DTT_ELEMS;
    const float* cw = conv_w + (size_t)l * DI * 4;
    const float* cb = conv_b + (size_t)l * DI;
    const float* db = dt_b  + (size_t)l * DI;
    const float* Dl = Dp    + (size_t)l * DI;

    // kick off async staging of x_proj / dt_proj weights into LDS;
    // they land while GEMM1's WMMA stream executes (ASYNCcnt-tracked).
    for (int i = tid; i < XPR_ELEMS / 8; i += 256)
      async_b128(&sXPR[i * 8], xprT + (size_t)i * 8);
    for (int i = tid; i < DTT_ELEMS / 8; i += 256)
      async_b128(&sDTT[i * 8], dtT + (size_t)i * 8);

    if (l + 1 < NL) __builtin_prefetch(inpT + LAYER_ELEMS + (size_t)tid * 64, 0, 1);

    // GEMM1: xz = h @ in_proj  (N=1024 -> 64 tiles, 8/wave as 4 dual passes)
    // A panel hoisted to registers; dual accumulators double B-loads in flight.
    {
      v16h areg[DM / 32];
#pragma unroll
      for (int k = 0; k < DM / 32; ++k)
        areg[k] = load_tile(sH, DM, 0, k * 32, lane);
      for (int jj = 0; jj < 4; ++jj) {
        int t0 = wave * 8 + jj * 2, t1 = t0 + 1;
        v8f acc0 = {}, acc1 = {};
#pragma unroll
        for (int k = 0; k < DM / 32; ++k) {
          v16h b0 = load_tile(inpT, DM, t0 * 16, k * 32, lane);
          v16h b1 = load_tile(inpT, DM, t1 * 16, k * 32, lane);
          acc0 = wmma16(areg[k], b0, acc0);
          acc1 = wmma16(areg[k], b1, acc1);
        }
#pragma unroll
        for (int s = 0; s < 2; ++s) {
          int t = s ? t1 : t0;
          v8f acc = s ? acc1 : acc0;
          int col = t * 16 + (lane & 15);          // tile-uniform side of split
          if (col < DI) {
            float c3 = cw[col * 4 + 3], cbv = cb[col];
#pragma unroll
            for (int v = 0; v < 8; ++v)
              sXC[(v + hi * 8) * DI + col] = (_Float16)silu_f(acc[v] * c3 + cbv);
          } else {
            int d = col - DI;
#pragma unroll
            for (int v = 0; v < 8; ++v)
              sZ[(v + hi * 8) * DI + d] = (_Float16)silu_f(acc[v]);
          }
        }
      }
    }
    wait_async0();        // staged weights resident before anyone reads them
    __syncthreads();

    // GEMM2: dbl = xc @ x_proj  (N=48 -> 3 tiles on waves 0..2, B from LDS)
    if (wave < 3) {
      v8f acc = {};
#pragma unroll 4
      for (int k0 = 0; k0 < DI; k0 += 32) {
        v16h a = load_tile(sXC, DI, 0, k0, lane);
        v16h b = load_tile(sXPR, DI, wave * 16, k0, lane);
        acc = wmma16(a, b, acc);
      }
      int col = wave * 16 + (lane & 15);
#pragma unroll
      for (int v = 0; v < 8; ++v) sDBL[(v + hi * 8) * NX + col] = acc[v];
    }
    __syncthreads();

    // bc[m] = dot(B_m, C_m); pad dt rows into K=32 A-tile
    if (tid < 16) {
      float s = 0.f;
#pragma unroll
      for (int i = 0; i < DSTATE; ++i)
        s += sDBL[tid * NX + DTR + i] * sDBL[tid * NX + DTR + DSTATE + i];
      sBC[tid] = s;
    }
    for (int i = tid; i < 16 * KDT; i += 256) {
      int m = i >> 5, k = i & 31;
      sDtA[i] = (k < DTR) ? (_Float16)sDBL[m * NX + k] : (_Float16)0.f;
    }
    __syncthreads();

    // GEMM3: dt = softplus(dtp @ dt_w + dt_b); fused y = xc*(dt*bc + D)*silu(z)
    for (int j = 0; j < 4; ++j) {
      int t = wave * 4 + j;
      v16h a = load_tile(sDtA, KDT, 0, 0, lane);
      v16h b = load_tile(sDTT, KDT, t * 16, 0, lane);
      v8f acc = {};
      acc = wmma16(a, b, acc);
      int col = t * 16 + (lane & 15);
      float dbv = db[col], Dv = Dl[col];
#pragma unroll
      for (int v = 0; v < 8; ++v) {
        int m = v + hi * 8;
        float dt = softplus_f(acc[v] + dbv);
        float yv = (float)sXC[m * DI + col] * (dt * sBC[m] + Dv)
                 * (float)sZ[m * DI + col];
        sY[m * DI + col] = (_Float16)yv;
      }
    }
    __syncthreads();

    // GEMM4: h = y @ out_proj  (N=256 -> 16 tiles, dual-tile per wave, K=512)
    {
      int t0 = wave * 2, t1 = t0 + 1;
      v8f acc0 = {}, acc1 = {};
#pragma unroll 4
      for (int k0 = 0; k0 < DI; k0 += 32) {
        v16h a  = load_tile(sY, DI, 0, k0, lane);
        v16h b0 = load_tile(outT, DI, t0 * 16, k0, lane);
        v16h b1 = load_tile(outT, DI, t1 * 16, k0, lane);
        acc0 = wmma16(a, b0, acc0);
        acc1 = wmma16(a, b1, acc1);
      }
      int c0 = t0 * 16 + (lane & 15), c1 = c0 + 16;
#pragma unroll
      for (int v = 0; v < 8; ++v) {
        int m = v + hi * 8;
        sH[m * DM + c0] = (_Float16)acc0[v];
        sH[m * DM + c1] = (_Float16)acc1[v];
      }
    }
    __syncthreads();
  }

  // out = h @ Wo + bo  (length-256 dot per row)
  if (tid < 16) {
    float s = 0.f;
#pragma unroll 8
    for (int d = 0; d < DM; ++d) s += (float)sH[tid * DM + d] * Wo[d];
    out[row0 + tid] = s + bo[0];
  }
}

// ---------------------------------------------------------------------------
extern "C" void kernel_launch(void* const* d_in, const int* in_sizes, int n_in,
                              void* d_out, int out_size, void* d_ws, size_t ws_size,
                              hipStream_t stream) {
  (void)in_sizes; (void)n_in; (void)out_size; (void)ws_size;
  const float* x    = (const float*)d_in[0];
  const float* Wi   = (const float*)d_in[1];
  const float* bi   = (const float*)d_in[2];
  const float* inp  = (const float*)d_in[3];
  const float* cw   = (const float*)d_in[4];
  const float* cb   = (const float*)d_in[5];
  const float* xpr  = (const float*)d_in[6];
  const float* dtw  = (const float*)d_in[7];
  const float* dtb  = (const float*)d_in[8];
  // d_in[9] = A_log: mathematically unused (h0 == 0 and L == 1 => dA*h0 == 0)
  const float* Dp   = (const float*)d_in[10];
  const float* outw = (const float*)d_in[11];
  const float* Wo   = (const float*)d_in[12];
  const float* bo   = (const float*)d_in[13];
  _Float16* ws = (_Float16*)d_ws;      // ~10.6 MB of f16 transposed weights
  float* out = (float*)d_out;

  prep_wiT <<<WIT_ELEMS  / 256, 256, 0, stream>>>(Wi,  ws);
  prep_inpT<<<dim3(INP_ELEMS  / 256, NL), 256, 0, stream>>>(inp,  ws);
  prep_xprT<<<dim3(XPR_ELEMS  / 256, NL), 256, 0, stream>>>(xpr,  ws);
  prep_dtT <<<dim3(DTT_ELEMS  / 256, NL), 256, 0, stream>>>(dtw,  ws);
  prep_outT<<<dim3(OUTP_ELEMS / 256, NL), 256, 0, stream>>>(outw, ws);

  mamba_fused<<<4096 / 16, 256, 0, stream>>>(x, bi, cw, cb, dtb, Dp, Wo, bo, ws, out);
}
